// PairContrastiveLoss_69174743269980
// MI455X (gfx1250) — compile-verified, matrix operand order checked by reading, not run on hardware
//
#include <hip/hip_runtime.h>

// ---------------------------------------------------------------------------
// PairContrastiveLoss on MI455X (gfx1250), wave32 + bf16 WMMA pipeline.
// Round 3: preload all LDS fragments per K-step so the 4 WMMAs issue
// back-to-back (no per-WMMA s_wait_dscnt 0 stalls). B-tile staged with
// GLOBAL_LOAD_ASYNC_TO_LDS_B128 (ASYNCcnt), A-gather software-pipelined,
// double-buffered LDS.
// ---------------------------------------------------------------------------

typedef __attribute__((ext_vector_type(16))) __bf16        v16bf;
typedef __attribute__((ext_vector_type(8)))  __bf16        v8bf;
typedef __attribute__((ext_vector_type(8)))  float         v8f;
typedef __attribute__((ext_vector_type(4)))  float         f32x4;
typedef __attribute__((ext_vector_type(4)))  unsigned int  u32x4;

union Frag16 { v16bf v; u32x4 q[2]; };   // 32B: one wmma bf16 A/B fragment
union Pack8  { v8bf  h; u32x4 q;    };   // 16B: 8 bf16

#define DDIM   768     // embedding dim (also K of patch GEMM: 3*16*16)
#define NPATCH 196     // 14x14 patches
#define RPAD   208     // padded to 13 tiles of 16
#define NBATCH 64
#define IMGW   224

// ---------------------------------------------------------------------------
// Kernel 0: convert conv weight to bf16 (flat layout already matches im2col K)
// ---------------------------------------------------------------------------
__global__ void k_wconv(const float* __restrict__ W, __bf16* __restrict__ Wb) {
    int i = blockIdx.x * 256 + threadIdx.x;   // exact grid: 768*768 elements
    Wb[i] = (__bf16)W[i];
}

// ---------------------------------------------------------------------------
// Kernel 1: fused im2col + patch-embedding GEMM (M=13312pad, N=768, K=768).
// Block: 256 thr (8 waves). Tile: 128M x 64N, K-step 32, double-buffered LDS.
// B tile goes global->LDS via async-to-LDS; A tile is gathered (im2col) into
// registers for step k+1 while step k's WMMAs run, then cvt+ds_store.
// ---------------------------------------------------------------------------
__global__ __launch_bounds__(256) void k_embed(const float* __restrict__ img,
                                               const __bf16* __restrict__ Wb,
                                               const float* __restrict__ bias,
                                               __bf16* __restrict__ P) {
    __shared__ __bf16 lA[2][128 * 32];   // 2 x 8 KB
    __shared__ __bf16 lB[2][64 * 32];    // 2 x 4 KB

    const int t    = threadIdx.x;
    const int w    = t >> 5;
    const int lane = t & 31;
    const int nl   = lane & 15;
    const int half = lane >> 4;
    const int g0   = blockIdx.x * 128;   // global padded-row base
    const int d0   = blockIdx.y * 64;    // output-channel base

    // --- A staging coords: thread gathers 16 contiguous floats of one row ---
    const int rloc = t >> 1;             // 0..127 tile row
    const int ks   = (t & 1) * 16;       // k sub-chunk 0 or 16
    const int g    = g0 + rloc;
    const int bi   = g / RPAD;
    const int p    = g - bi * RPAD;
    const bool valid = (p < NPATCH);
    const int ph = p / 14, pw = p - (p / 14) * 14;
    const float* imgBase = img + (size_t)bi * 3 * IMGW * IMGW
                               + (size_t)(ph * 16) * IMGW + pw * 16;

    // --- B staging coords: thread owns one 16B chunk of the weight tile ---
    const int dl = t >> 2;               // 0..63
    const int kb = (t & 3) * 8;
    const __bf16* wRow = Wb + (size_t)(d0 + dl) * DDIM + kb;

    v8f acc[4] = {};

    // ---------------- prologue: stage K-step 0 into buffer 0 ----------------
    {
        // B: async global -> LDS (ASYNCcnt)
        unsigned ldsOff = (unsigned)(size_t)(&lB[0][dl * 32 + kb]);
        const __bf16* gsrc = wRow;
        asm volatile("global_load_async_to_lds_b128 %0, %1, off"
                     :: "v"(ldsOff), "v"(gsrc) : "memory");
        // A: gather + fp32->bf16 + ds_store
        Pack8 p0, p1;
        if (valid) {
            const int rr = ks >> 4;                 // kc=0 -> channel 0
            const float* src = imgBase + (size_t)rr * IMGW;
            f32x4 v0 = ((const f32x4*)src)[0];
            f32x4 v1 = ((const f32x4*)src)[1];
            f32x4 v2 = ((const f32x4*)src)[2];
            f32x4 v3 = ((const f32x4*)src)[3];
#pragma unroll
            for (int e = 0; e < 4; ++e) {
                p0.h[e]     = (__bf16)v0[e];
                p0.h[4 + e] = (__bf16)v1[e];
                p1.h[e]     = (__bf16)v2[e];
                p1.h[4 + e] = (__bf16)v3[e];
            }
        } else {
            u32x4 z = {0u, 0u, 0u, 0u};
            p0.q = z; p1.q = z;
        }
        u32x4* dst = (u32x4*)(&lA[0][rloc * 32 + ks]);
        dst[0] = p0.q;
        dst[1] = p1.q;
        asm volatile("s_wait_asynccnt 0x0" ::: "memory");
    }
    __syncthreads();

    // ---------------- main loop: compute cur, stage nxt ----------------
    for (int kc = 0; kc < DDIM; kc += 32) {
        const int cur = (kc >> 5) & 1;
        const int nxt = cur ^ 1;
        const bool hasNext = (kc + 32) < DDIM;

        f32x4 v0, v1, v2, v3;
        if (hasNext) {
            // issue async B for next step into the other buffer
            unsigned ldsOff = (unsigned)(size_t)(&lB[nxt][dl * 32 + kb]);
            const __bf16* gsrc = wRow + (kc + 32);
            asm volatile("global_load_async_to_lds_b128 %0, %1, off"
                         :: "v"(ldsOff), "v"(gsrc) : "memory");
            // issue A-gather loads for next step (consumed after the WMMAs)
            if (valid) {
                const int kp = kc + 32 + ks;
                const int c  = kp >> 8;
                const int rr = (kp >> 4) & 15;
                const float* src = imgBase + (size_t)c * IMGW * IMGW + (size_t)rr * IMGW;
                v0 = ((const f32x4*)src)[0];
                v1 = ((const f32x4*)src)[1];
                v2 = ((const f32x4*)src)[2];
                v3 = ((const f32x4*)src)[3];
            }
        }

        // ---- compute from current buffer ----
        // Preload A fragment and all 4 B fragments first, then issue the
        // 4 WMMAs back-to-back (avoids a DScnt==0 stall in front of each).
        {
            const __bf16* aFragP = &lA[cur][(w * 16 + nl) * 32 + half * 8];
            Frag16 a;
            a.q[0] = *(const u32x4*)(aFragP);
            a.q[1] = *(const u32x4*)(aFragP + 16);
            Frag16 bm[4];
#pragma unroll
            for (int j = 0; j < 4; ++j) {
                const __bf16* bFragP = &lB[cur][(j * 16 + nl) * 32 + half * 8];
                bm[j].q[0] = *(const u32x4*)(bFragP);
                bm[j].q[1] = *(const u32x4*)(bFragP + 16);
            }
#pragma unroll
            for (int j = 0; j < 4; ++j) {
                acc[j] = __builtin_amdgcn_wmma_f32_16x16x32_bf16(
                    false, a.v, false, bm[j].v, (short)0, acc[j], false, false);
            }
        }

        // ---- finish staging A into the other buffer ----
        if (hasNext) {
            Pack8 p0, p1;
            if (valid) {
#pragma unroll
                for (int e = 0; e < 4; ++e) {
                    p0.h[e]     = (__bf16)v0[e];
                    p0.h[4 + e] = (__bf16)v1[e];
                    p1.h[e]     = (__bf16)v2[e];
                    p1.h[4 + e] = (__bf16)v3[e];
                }
            } else {
                u32x4 z = {0u, 0u, 0u, 0u};
                p0.q = z; p1.q = z;
            }
            u32x4* dst = (u32x4*)(&lA[nxt][rloc * 32 + ks]);
            dst[0] = p0.q;
            dst[1] = p1.q;
            asm volatile("s_wait_asynccnt 0x0" ::: "memory");
        }
        __syncthreads();
    }

    // ---- epilogue: + bias, store bf16 (C layout: VGPR i, lane-half -> M) ----
#pragma unroll
    for (int j = 0; j < 4; ++j) {
        const int ng = d0 + j * 16 + nl;
        const float bv = bias[ng];
#pragma unroll
        for (int i = 0; i < 8; ++i) {
            const int m = w * 16 + i + 8 * half;
            const float val = acc[j][i] + bv;
            P[(size_t)(g0 + m) * DDIM + ng] = (__bf16)val;
        }
    }
}

// ---------------------------------------------------------------------------
// Kernel 2: row L2 norms of the bf16 embeddings (one wave per row).
// ---------------------------------------------------------------------------
__global__ __launch_bounds__(256) void k_norm(const __bf16* __restrict__ P,
                                              float* __restrict__ nrm) {
    const int w    = threadIdx.x >> 5;
    const int lane = threadIdx.x & 31;
    const int row  = blockIdx.x * 8 + w;
    const __bf16* rp = P + (size_t)row * DDIM + lane * 8;
    float s = 0.f;
#pragma unroll
    for (int c = 0; c < 3; ++c) {
        Pack8 pk;
        pk.q = *(const u32x4*)(rp + c * 256);
#pragma unroll
        for (int e = 0; e < 8; ++e) {
            const float v = (float)pk.h[e];
            s = fmaf(v, v, s);
        }
    }
#pragma unroll
    for (int m = 16; m; m >>= 1) s += __shfl_xor(s, m, 32);
    if (lane == 0) nrm[row] = sqrtf(s);
}

// ---------------------------------------------------------------------------
// Kernel 3: per-batch 196x196 cosine-sim dots via WMMA + masked exp-sum.
// Grid: (64 batches, 13 M-tiles). Waves stride over the 13 N-tiles.
// Fragments loaded directly from global (embeddings are L2-resident).
// ---------------------------------------------------------------------------
__global__ __launch_bounds__(256) void k_loss(const __bf16* __restrict__ Px,
                                              const __bf16* __restrict__ Py,
                                              const float* __restrict__ nx,
                                              const float* __restrict__ ny,
                                              float* __restrict__ sb2) {
    const int b  = blockIdx.x;
    const int mt = blockIdx.y;
    const int t    = threadIdx.x;
    const int w    = t >> 5;
    const int lane = t & 31;
    const int nl   = lane & 15;
    const int half = lane >> 4;
    const size_t rbase = (size_t)b * RPAD;
    const int m0 = mt * 16;

    const __bf16* pA = Px + (rbase + m0 + nl) * DDIM + half * 8;

    float local = 0.f;
    for (int nt = w; nt < 13; nt += 8) {
        const int n0 = nt * 16;
        const __bf16* pB = Py + (rbase + n0 + nl) * DDIM + half * 8;
        v8f acc = {};
        for (int kc = 0; kc < DDIM; kc += 32) {
            Frag16 a, bm;
            a.q[0]  = *(const u32x4*)(pA + kc);
            a.q[1]  = *(const u32x4*)(pA + kc + 16);
            bm.q[0] = *(const u32x4*)(pB + kc);
            bm.q[1] = *(const u32x4*)(pB + kc + 16);
            acc = __builtin_amdgcn_wmma_f32_16x16x32_bf16(
                false, a.v, false, bm.v, (short)0, acc, false, false);
        }
        const int n = n0 + nl;
        const float nyv = ny[rbase + n];
#pragma unroll
        for (int i = 0; i < 8; ++i) {
            const int m = m0 + i + 8 * half;
            if (m < NPATCH && n < NPATCH && m != n) {
                const float denom = fmaxf(nx[rbase + m] * nyv, 1e-8f);
                local += __expf(2.0f * (acc[i] / denom));
            }
        }
    }

    __shared__ float red[256];
    red[t] = local;
    __syncthreads();
#pragma unroll
    for (int s = 128; s > 0; s >>= 1) {
        if (t < s) red[t] += red[t + s];
        __syncthreads();
    }
    if (t == 0) sb2[b * 13 + mt] = red[0];
}

// ---------------------------------------------------------------------------
// Kernel 4: out = mean_b log( sum_mt sb2[b][mt] )
// ---------------------------------------------------------------------------
__global__ void k_final(const float* __restrict__ sb2, float* __restrict__ out) {
    __shared__ float red[64];
    const int t = threadIdx.x;   // 64 threads
    float s = 0.f;
    for (int i = 0; i < 13; ++i) s += sb2[t * 13 + i];
    red[t] = logf(s);
    __syncthreads();
    for (int k = 32; k > 0; k >>= 1) {
        if (t < k) red[t] += red[t + k];
        __syncthreads();
    }
    if (t == 0) out[0] = red[0] * (1.0f / 64.0f);
}

// ---------------------------------------------------------------------------
// Launch
// ---------------------------------------------------------------------------
extern "C" void kernel_launch(void* const* d_in, const int* in_sizes, int n_in,
                              void* d_out, int out_size, void* d_ws, size_t ws_size,
                              hipStream_t stream) {
    (void)in_sizes; (void)n_in; (void)out_size; (void)ws_size;

    const float* x    = (const float*)d_in[0];   // (64,3,224,224)
    const float* y    = (const float*)d_in[1];   // (64,3,224,224)
    const float* W    = (const float*)d_in[2];   // (768,3,16,16)
    const float* bias = (const float*)d_in[3];   // (768,)
    float* out = (float*)d_out;

    char* ws = (char*)d_ws;
    __bf16* Wb  = (__bf16*)(ws);                          // 1.2 MB
    __bf16* Px  = (__bf16*)(ws + ((size_t)2  << 20));     // 19.5 MB (64*208*768 bf16)
    __bf16* Py  = (__bf16*)(ws + ((size_t)22 << 20));     // 19.5 MB
    float*  nx  = (float*)(ws + ((size_t)42 << 20));              // 52 KB
    float*  ny  = (float*)(ws + ((size_t)42 << 20) + (64 << 10)); // 52 KB
    float*  sb2 = (float*)(ws + ((size_t)42 << 20) + (128 << 10));// 64*13 f32

    // W -> bf16
    k_wconv<<<dim3(768 * 768 / 256), dim3(256), 0, stream>>>(W, Wb);

    // Patch embeddings (fused im2col GEMM), rows padded per batch to 208.
    dim3 gEmbed(NBATCH * RPAD / 128, DDIM / 64);
    k_embed<<<gEmbed, dim3(256), 0, stream>>>(x, Wb, bias, Px);
    k_embed<<<gEmbed, dim3(256), 0, stream>>>(y, Wb, bias, Py);

    // Row norms of bf16 embeddings (self-consistent with the WMMA dots).
    k_norm<<<dim3(NBATCH * RPAD / 8), dim3(256), 0, stream>>>(Px, nx);
    k_norm<<<dim3(NBATCH * RPAD / 8), dim3(256), 0, stream>>>(Py, ny);

    // Per-batch similarity + masked exp reduction (deterministic partials).
    k_loss<<<dim3(NBATCH, 13), dim3(256), 0, stream>>>(Px, Py, nx, ny, sb2);

    // Final log-mean.
    k_final<<<dim3(1), dim3(64), 0, stream>>>(sb2, out);
}
